// GNNMultiview_31568009626150
// MI455X (gfx1250) — matrix operand order repaired and checked by reading, not compiled, not run on hardware
//
#include <hip/hip_runtime.h>
#include <hip/hip_bf16.h>

// ---------------------------------------------------------------------------
// Problem constants (from reference)
// ---------------------------------------------------------------------------
#define NODES   144          // B*CH = 16*9
#define NCH     9
#define NB      16
#define TSLEN   3000
#define HID     512
#define ODIM    64
#define DLAT    2112         // 33*64
#define NEDGE   1152         // nodes*(ch-1)
#define KMSG    4224         // 2*DLAT
#define NROUND  3
#define LDS_STR 40           // padded K-stride (ushorts) for 32-wide tiles

typedef __attribute__((ext_vector_type(16))) __bf16 v16bf;
typedef __attribute__((ext_vector_type(8)))  float  v8f;

// ---------------------------------------------------------------------------
// Helpers
// ---------------------------------------------------------------------------
__device__ __forceinline__ unsigned short f2bf(float f) {
    unsigned int u = __float_as_uint(f);
    u += 0x7FFFu + ((u >> 16) & 1u);           // round-to-nearest-even
    return (unsigned short)(u >> 16);
}

__device__ __forceinline__ float gelu_tanh(float x) {
    const float c = 0.7978845608028654f;       // sqrt(2/pi)
    float t = tanhf(c * (x + 0.044715f * x * x * x));
    return 0.5f * x * (1.0f + t);
}

__device__ __forceinline__ v8f wmma_bf16(v16bf a, v16bf b, v8f c) {
    return __builtin_amdgcn_wmma_f32_16x16x32_bf16(
        /*neg_a=*/false, a, /*neg_b=*/false, b,
        /*c_mod=*/(short)0, c, /*reuse_a=*/false, /*reuse_b=*/false);
}

// A fragment: 16x32 bf16, row-major LDS tile [M][K], stride LDS_STR ushorts.
// Lane l: M = l&15 ; khalf = l>>4.  VGPR v(0..3): K = 8*khalf + 2v,+1 ;
// v(4..7): K = 16 + 8*khalf + 2(v-4),+1.  Pairs contiguous -> b128 ds reads.
__device__ __forceinline__ v16bf load_a_frag(const unsigned short* Alds,
                                             int m_off, int lane) {
    union { v16bf v; unsigned int u[8]; } A;
    int am = lane & 15, kh = lane >> 4;
    const unsigned int* row =
        (const unsigned int*)(Alds + (size_t)(m_off + am) * LDS_STR);
    int b = kh * 4;
#pragma unroll
    for (int i = 0; i < 4; ++i) A.u[i]     = row[b + i];
#pragma unroll
    for (int i = 0; i < 4; ++i) A.u[4 + i] = row[8 + b + i];
    return A.v;
}

// B fragment: 32x16 bf16 from LDS tile stored TRANSPOSED [N][K], stride LDS_STR.
// Lane l: N = l&15 ; kgrp = l>>4.  VGPR v(0..3): K = 8*kgrp + 2v,+1 ; v(4..7): K = 16 + 8*kgrp + 2(v-4),+1.
__device__ __forceinline__ v16bf load_b_frag(const unsigned short* BldsT,
                                             int n_off, int lane) {
    union { v16bf v; unsigned int u[8]; } B;
    int bn = lane & 15, kg = lane >> 4;
    const unsigned int* row =
        (const unsigned int*)(BldsT + (size_t)(n_off + bn) * LDS_STR);
#pragma unroll
    for (int i = 0; i < 8; ++i) B.u[i] = row[(i >> 2) * 8 + kg * 4 + (i & 3)];
    return B.v;
}

// ---------------------------------------------------------------------------
// K0: f32 -> bf16 weight packing
// ---------------------------------------------------------------------------
__global__ __launch_bounds__(256) void cvt_f32_bf16(
    const float* __restrict__ src, unsigned short* __restrict__ dst, int n) {
    int i = blockIdx.x * 256 + threadIdx.x;
    if (i < n) dst[i] = f2bf(src[i]);
}

// ---------------------------------------------------------------------------
// K1: first conv (1->512, k=3, stride=3, pad=1) + GroupNorm + GELU, -> bf16
// grid (512, NODES), block 256
// ---------------------------------------------------------------------------
__global__ __launch_bounds__(256) void conv_first_gn_gelu(
    const float* __restrict__ x,      // [NODES][3000]
    const float* __restrict__ w,      // [512][3]
    const float* __restrict__ bias,
    const float* __restrict__ gw, const float* __restrict__ gb,
    unsigned short* __restrict__ out) // [NODES][512][1000] bf16
{
    const int c = blockIdx.x, node = blockIdx.y;
    __shared__ float yv[1000];
    __shared__ float red[256], red2[256];
    const float* xr = x + (size_t)node * TSLEN;
    const float w0 = w[c * 3 + 0], w1 = w[c * 3 + 1], w2 = w[c * 3 + 2];
    const float b = bias[c];
    float s = 0.f, s2 = 0.f;
    for (int t = threadIdx.x; t < 1000; t += 256) {
        int p = 3 * t - 1;
        int pc = p < 0 ? 0 : p;
        float x0 = xr[pc];                       // branchless left pad
        float v = b + w1 * xr[p + 1] + w2 * xr[p + 2];
        v += (p >= 0) ? w0 * x0 : 0.0f;
        yv[t] = v; s += v; s2 += v * v;
    }
    red[threadIdx.x] = s; red2[threadIdx.x] = s2;
    __syncthreads();
    for (int o = 128; o > 0; o >>= 1) {
        if (threadIdx.x < o) {
            red[threadIdx.x]  += red[threadIdx.x + o];
            red2[threadIdx.x] += red2[threadIdx.x + o];
        }
        __syncthreads();
    }
    const float mu  = red[0] * (1.0f / 1000.0f);
    const float var = red2[0] * (1.0f / 1000.0f) - mu * mu;
    const float inv = rsqrtf(var + 1e-5f);
    const float g = gw[c], gbb = gb[c];
    unsigned short* o = out + ((size_t)node * HID + c) * 1000;
    for (int t = threadIdx.x; t < 1000; t += 256)
        o[t] = f2bf(gelu_tanh((yv[t] - mu) * inv * g + gbb));
}

// ---------------------------------------------------------------------------
// K2: conv-as-GEMM via WMMA, double-buffered LDS pipeline.
// Y[node][co][t] = sum_K W[co][K] * col[K][t], K = 2*cin+k,
// col[K][t] = X[cin][2t-1+k] (zero-padded).
// grid ((Lout+31)/32, ceil(Cout/128), NODES), block 256 = 8 waves.
// Wave: 16 rows x 32 cols (two accumulators, A-fragment reuse).
// ---------------------------------------------------------------------------
__global__ __launch_bounds__(256) void gemm_conv(
    const unsigned short* __restrict__ Xbf, // [NODES][512][Lin] bf16
    const unsigned short* __restrict__ Wbf, // [Cout][1024] bf16
    float* __restrict__ Y,                  // [NODES][Cout][Lout]
    int Lin, int Lout, int Cout)
{
    __shared__ __align__(16) unsigned short Alds[2][128 * LDS_STR];
    __shared__ __align__(16) unsigned short Blds[2][32 * LDS_STR]; // [t][K]
    const int node = blockIdx.z;
    const int m0 = blockIdx.y * 128;
    const int n0 = blockIdx.x * 32;
    const int tid = threadIdx.x, wave = tid >> 5, lane = tid & 31;
    const unsigned short* Xrow = Xbf + (size_t)node * HID * Lin;
    const int KTOT = 1024;

    // per-thread load roles
    const int arow = tid >> 1, aoff = (tid & 1) * 16;   // A: 128x32, 2 uint4
    const int bk = tid >> 3,  bt0 = (tid & 7) * 4;      // B: 32(K)x32(t)

    uint4 ra0, ra1; unsigned short rb[4];
    auto fetch = [&](int kb) {
        int gm = m0 + arow;
        ra0 = make_uint4(0, 0, 0, 0); ra1 = ra0;
        if (gm < Cout) {
            const uint4* s = (const uint4*)(Wbf + (size_t)gm * KTOT + kb + aoff);
            ra0 = s[0]; ra1 = s[1];
            if (kb + 32 < KTOT)
                __builtin_prefetch(Wbf + (size_t)gm * KTOT + kb + 32 + aoff, 0, 0);
        }
        int c  = (kb + bk) >> 1;
        int kk = (kb + bk) & 1;
        const unsigned short* xc = Xrow + (size_t)c * Lin;
#pragma unroll
        for (int i = 0; i < 4; ++i) {                    // branchless pad
            int pos = 2 * (n0 + bt0 + i) - 1 + kk;
            bool inb = (pos >= 0) && (pos < Lin);
            unsigned short v = xc[inb ? pos : 0];
            rb[i] = inb ? v : (unsigned short)0;
        }
    };
    auto stash = [&](int buf) {
        uint4* d = (uint4*)(Alds[buf] + arow * LDS_STR + aoff);
        d[0] = ra0; d[1] = ra1;
#pragma unroll
        for (int i = 0; i < 4; ++i)
            Blds[buf][(bt0 + i) * LDS_STR + bk] = rb[i];
    };

    v8f acc0 = {}, acc1 = {};
    const bool mok = (m0 + wave * 16) < Cout;           // wave-uniform

    fetch(0); stash(0);
    __syncthreads();
    const int nk = KTOT / 32;
    for (int i = 0; i < nk; ++i) {
        const int cur = i & 1;
        const bool more = (i + 1) < nk;
        if (more) fetch((i + 1) * 32);
        if (mok) {                                      // EXEC all-1s inside
            v16bf a  = load_a_frag(Alds[cur], wave * 16, lane);
            v16bf b0 = load_b_frag(Blds[cur], 0, lane);
            v16bf b1 = load_b_frag(Blds[cur], 16, lane);
            acc0 = wmma_bf16(a, b0, acc0);
            acc1 = wmma_bf16(a, b1, acc1);
        }
        if (more) stash(cur ^ 1);
        __syncthreads();
    }

    if (mok) {
        const int mbase = m0 + wave * 16 + ((lane >> 4) << 3);
        const int ln = n0 + (lane & 15);
        float* base = Y + (size_t)node * Cout * Lout + (size_t)mbase * Lout;
        if (ln < Lout) {
            float* p = base + ln;
#pragma unroll
            for (int r = 0; r < 8; ++r) p[(size_t)r * Lout] = acc0[r];
        }
        if (ln + 16 < Lout) {
            float* p = base + ln + 16;
#pragma unroll
            for (int r = 0; r < 8; ++r) p[(size_t)r * Lout] = acc1[r];
        }
    }
}

// ---------------------------------------------------------------------------
// K3: GroupNorm (per node,channel over time) + GELU; bias folded in.
// grid (C, NODES), block 256. Writes bf16 act; optionally fp32 (latents).
// ---------------------------------------------------------------------------
__global__ __launch_bounds__(256) void gn_gelu(
    const float* __restrict__ Y, const float* __restrict__ bias,
    const float* __restrict__ gw, const float* __restrict__ gb,
    unsigned short* __restrict__ out, float* __restrict__ outF,
    int C, int L)
{
    const int c = blockIdx.x, node = blockIdx.y;
    const float* y = Y + ((size_t)node * C + c) * L;
    __shared__ float red[256], red2[256];
    const float b = bias[c];
    float s = 0.f, s2 = 0.f;
    for (int t = threadIdx.x; t < L; t += 256) {
        float v = y[t] + b; s += v; s2 += v * v;
    }
    red[threadIdx.x] = s; red2[threadIdx.x] = s2;
    __syncthreads();
    for (int o = 128; o > 0; o >>= 1) {
        if (threadIdx.x < o) {
            red[threadIdx.x]  += red[threadIdx.x + o];
            red2[threadIdx.x] += red2[threadIdx.x + o];
        }
        __syncthreads();
    }
    const float invL = 1.0f / (float)L;
    const float mu  = red[0] * invL;
    const float var = red2[0] * invL - mu * mu;
    const float inv = rsqrtf(var + 1e-5f);
    const float g = gw[c], gbb = gb[c];
    unsigned short* o = out + ((size_t)node * C + c) * L;
    if (outF) {
        float* of = outF + ((size_t)node * C + c) * L;
        for (int t = threadIdx.x; t < L; t += 256) {
            float v = gelu_tanh((y[t] + b - mu) * inv * g + gbb);
            o[t] = f2bf(v); of[t] = v;
        }
    } else {
        for (int t = threadIdx.x; t < L; t += 256)
            o[t] = f2bf(gelu_tanh((y[t] + b - mu) * inv * g + gbb));
    }
}

// ---------------------------------------------------------------------------
// K4: gather edge features  E[e] = [lat_bf[from] | lat_bf[to]]
// ---------------------------------------------------------------------------
__global__ __launch_bounds__(256) void gather_edges(
    const unsigned short* __restrict__ latB, unsigned short* __restrict__ E)
{
    const int e = blockIdx.x;
    const int s = e / 72, r = e % 72, i = r >> 3, jr = r & 7;
    const int j = jr + (jr >= i ? 1 : 0);
    const int from = s * NCH + i, to = s * NCH + j;
    const uint4* fa = (const uint4*)(latB + (size_t)from * DLAT);
    const uint4* fb = (const uint4*)(latB + (size_t)to   * DLAT);
    uint4* dst = (uint4*)(E + (size_t)e * KMSG);
    for (int q = threadIdx.x; q < DLAT / 8; q += 256) {
        dst[q] = fa[q];
        dst[DLAT / 8 + q] = fb[q];
    }
}

// ---------------------------------------------------------------------------
// K5: message GEMM  M = tanh(E @ W + b)  [1152 x 4224]x[4224 x 2112]
// Double-buffered. grid (2112/32, 1152/128), block 256 = 8 waves.
// Wave: 16 rows x 32 cols (two accumulators, A-fragment reuse).
// ---------------------------------------------------------------------------
__global__ __launch_bounds__(256) void gemm_msg(
    const unsigned short* __restrict__ E,   // [1152][4224]
    const unsigned short* __restrict__ Wb,  // [4224][2112]
    const float* __restrict__ bias,         // [2112]
    float* __restrict__ M)                  // [1152][2112]
{
    __shared__ __align__(16) unsigned short Alds[2][128 * LDS_STR];
    __shared__ __align__(16) unsigned short Blds[2][32 * LDS_STR]; // [n][k]
    const int n0 = blockIdx.x * 32, m0 = blockIdx.y * 128;
    const int tid = threadIdx.x, wave = tid >> 5, lane = tid & 31;

    const int arow = tid >> 1, aoff = (tid & 1) * 16;   // A: 128x32, 2 uint4
    const int bk = tid >> 3,  bc0 = (tid & 7) * 4;      // B: 32(K)x32(n)

    uint4 ra0, ra1; unsigned int rb0, rb1;
    auto fetch = [&](int kb) {
        const uint4* s = (const uint4*)(E + (size_t)(m0 + arow) * KMSG + kb + aoff);
        ra0 = s[0]; ra1 = s[1];
        const unsigned short* wsrc = Wb + (size_t)(kb + bk) * DLAT + n0 + bc0;
        rb0 = *(const unsigned int*)(wsrc);
        rb1 = *(const unsigned int*)(wsrc + 2);
        if (kb + 32 < KMSG)
            __builtin_prefetch(Wb + (size_t)(kb + 32 + bk) * DLAT + n0 + bc0, 0, 0);
    };
    auto stash = [&](int buf) {
        uint4* d = (uint4*)(Alds[buf] + arow * LDS_STR + aoff);
        d[0] = ra0; d[1] = ra1;
        Blds[buf][(bc0 + 0) * LDS_STR + bk] = (unsigned short)(rb0 & 0xffff);
        Blds[buf][(bc0 + 1) * LDS_STR + bk] = (unsigned short)(rb0 >> 16);
        Blds[buf][(bc0 + 2) * LDS_STR + bk] = (unsigned short)(rb1 & 0xffff);
        Blds[buf][(bc0 + 3) * LDS_STR + bk] = (unsigned short)(rb1 >> 16);
    };

    v8f acc0 = {}, acc1 = {};
    fetch(0); stash(0);
    __syncthreads();
    const int nk = KMSG / 32;
    for (int i = 0; i < nk; ++i) {
        const int cur = i & 1;
        const bool more = (i + 1) < nk;
        if (more) fetch((i + 1) * 32);
        {
            v16bf a  = load_a_frag(Alds[cur], wave * 16, lane);
            v16bf b0 = load_b_frag(Blds[cur], 0, lane);
            v16bf b1 = load_b_frag(Blds[cur], 16, lane);
            acc0 = wmma_bf16(a, b0, acc0);
            acc1 = wmma_bf16(a, b1, acc1);
        }
        if (more) stash(cur ^ 1);
        __syncthreads();
    }

    const int mbase = m0 + wave * 16 + ((lane >> 4) << 3);
    const int n = n0 + (lane & 15);
    const float bv0 = bias[n], bv1 = bias[n + 16];
    float* p = M + (size_t)mbase * DLAT + n;
#pragma unroll
    for (int r = 0; r < 8; ++r) {
        p[(size_t)r * DLAT]      = tanhf(acc0[r] + bv0);
        p[(size_t)r * DLAT + 16] = tanhf(acc1[r] + bv1);
    }
}

// ---------------------------------------------------------------------------
// K6: scatter-add messages into latents (8 statically-known edges per node).
// ---------------------------------------------------------------------------
__global__ __launch_bounds__(256) void scatter_msgs(
    const float* __restrict__ M, float* __restrict__ latF,
    unsigned short* __restrict__ latB)
{
    const int idx = blockIdx.x * 256 + threadIdx.x;
    if (idx >= NODES * DLAT) return;
    const int n = idx / DLAT, d = idx % DLAT;
    const int s = n / NCH, j = n % NCH;
    float acc = latF[idx];
#pragma unroll
    for (int i = 0; i < NCH; ++i) {
        if (i == j) continue;
        int jr = (j > i) ? j - 1 : j;
        int e = s * 72 + i * 8 + jr;
        acc += M[(size_t)e * DLAT + d];
    }
    latF[idx] = acc;
    latB[idx] = f2bf(acc);
}

// ---------------------------------------------------------------------------
// K7: per-sample node sum  Z[s] = sum_ch latents
// ---------------------------------------------------------------------------
__global__ __launch_bounds__(256) void reduce_nodes(
    const float* __restrict__ latF, unsigned short* __restrict__ Zb)
{
    const int idx = blockIdx.x * 256 + threadIdx.x;
    if (idx >= NB * DLAT) return;
    const int s = idx / DLAT, d = idx % DLAT;
    float a = 0.f;
#pragma unroll
    for (int c = 0; c < NCH; ++c)
        a += latF[(size_t)(s * NCH + c) * DLAT + d];
    Zb[idx] = f2bf(a);
}

// ---------------------------------------------------------------------------
// K8: readout GEMM  out = tanh(Z @ Wr + br)  [16 x 2112]x[2112 x 2112]
// grid ceil(2112/128), block 256 = 8 waves, one 16x16 tile each.
// ---------------------------------------------------------------------------
__global__ __launch_bounds__(256) void gemm_readout(
    const unsigned short* __restrict__ Zb,  // [16][2112]
    const unsigned short* __restrict__ Wb,  // [2112][2112]
    const float* __restrict__ bias, float* __restrict__ out)
{
    __shared__ __align__(16) unsigned short Alds[16 * LDS_STR];
    __shared__ __align__(16) unsigned short Blds[128 * LDS_STR]; // [n][k]
    const int n0 = blockIdx.x * 128;
    const int tid = threadIdx.x, wave = tid >> 5, lane = tid & 31;
    const int nsub = wave * 16;
    v8f acc = {};
    for (int kb = 0; kb < DLAT; kb += 32) {
        if (tid < 64) { // A tile 16x32
            int row = tid >> 2, cc = (tid & 3) * 8;
            *(uint4*)(Alds + row * LDS_STR + cc) =
                *(const uint4*)(Zb + (size_t)row * DLAT + kb + cc);
        }
        { // B tile 32(K) x 128(n), transposed, branchless zero-fill past DLAT
            int k = tid >> 3, c0 = (tid & 7) * 16;
            const unsigned short* src = Wb + (size_t)(kb + k) * DLAT;
#pragma unroll
            for (int q = 0; q < 8; ++q) {
                int col = n0 + c0 + 2 * q;
                bool inb = col < DLAT;                  // DLAT even -> col+1 ok
                unsigned int p = *(const unsigned int*)(src + (inb ? col : 0));
                p = inb ? p : 0u;
                Blds[(c0 + 2 * q + 0) * LDS_STR + k] = (unsigned short)(p & 0xffff);
                Blds[(c0 + 2 * q + 1) * LDS_STR + k] = (unsigned short)(p >> 16);
            }
        }
        __syncthreads();
        v16bf a = load_a_frag(Alds, 0, lane);
        v16bf b = load_b_frag(Blds, nsub, lane);
        acc = wmma_bf16(a, b, acc);
        __syncthreads();
    }
    const int m = (lane >> 4) << 3;
    const int n = n0 + nsub + (lane & 15);
    if (n < DLAT) {
        const float bv = bias[n];
        float* p = out + (size_t)m * DLAT + n;
#pragma unroll
        for (int r = 0; r < 8; ++r) p[(size_t)r * DLAT] = tanhf(acc[r] + bv);
    }
}

// ---------------------------------------------------------------------------
// Host orchestration
// ---------------------------------------------------------------------------
extern "C" void kernel_launch(void* const* d_in, const int* in_sizes, int n_in,
                              void* d_out, int out_size, void* d_ws, size_t ws_size,
                              hipStream_t stream) {
    const float* x            = (const float*)d_in[0];
    const float* conv_w_first = (const float*)d_in[1];
    const float* conv_b_first = (const float*)d_in[2];
    const float* gn_w_first   = (const float*)d_in[3];
    const float* gn_b_first   = (const float*)d_in[4];
    const float* conv_w_mid   = (const float*)d_in[5];
    const float* conv_b_mid   = (const float*)d_in[6];
    const float* gn_w_mid     = (const float*)d_in[7];
    const float* gn_b_mid     = (const float*)d_in[8];
    const float* conv_w_last  = (const float*)d_in[9];
    const float* conv_b_last  = (const float*)d_in[10];
    const float* gn_w_last    = (const float*)d_in[11];
    const float* gn_b_last    = (const float*)d_in[12];
    const float* msg_w        = (const float*)d_in[13];
    const float* msg_b        = (const float*)d_in[14];
    const float* readout_w    = (const float*)d_in[15];
    const float* readout_b    = (const float*)d_in[16];

    // workspace carve-up (assumes ws_size ~>= 460 MB)
    char* ws = (char*)d_ws;
    size_t off = 0;
    auto carve = [&](size_t bytes) -> char* {
        char* p = ws + off;
        off = (off + bytes + 255) & ~(size_t)255;
        return p;
    };
    unsigned short* actA   = (unsigned short*)carve((size_t)NODES * HID * 1000 * 2);
    unsigned short* actB   = (unsigned short*)carve((size_t)NODES * HID * 501 * 2);
    float*          Ybuf   = (float*)carve((size_t)NODES * HID * 501 * 4);
    float*          latF   = (float*)carve((size_t)NODES * DLAT * 4);
    unsigned short* latB   = (unsigned short*)carve((size_t)NODES * DLAT * 2);
    unsigned short* wmidB  = (unsigned short*)carve((size_t)4 * HID * 1024 * 2);
    unsigned short* wlastB = (unsigned short*)carve((size_t)ODIM * 1024 * 2);
    unsigned short* msgwB  = (unsigned short*)carve((size_t)NROUND * KMSG * DLAT * 2);
    unsigned short* rwB    = (unsigned short*)carve((size_t)DLAT * DLAT * 2);
    unsigned short* Ebuf   = (unsigned short*)carve((size_t)NEDGE * KMSG * 2);
    float*          Mbuf   = (float*)carve((size_t)NEDGE * DLAT * 4);
    unsigned short* Zb     = (unsigned short*)carve((size_t)NB * DLAT * 2);

    // weight packing f32 -> bf16
    {
        int n1 = 4 * HID * 1024;
        cvt_f32_bf16<<<(n1 + 255) / 256, 256, 0, stream>>>(conv_w_mid, wmidB, n1);
        int n2 = ODIM * 1024;
        cvt_f32_bf16<<<(n2 + 255) / 256, 256, 0, stream>>>(conv_w_last, wlastB, n2);
        int n3 = NROUND * KMSG * DLAT;
        cvt_f32_bf16<<<(n3 + 255) / 256, 256, 0, stream>>>(msg_w, msgwB, n3);
        int n4 = DLAT * DLAT;
        cvt_f32_bf16<<<(n4 + 255) / 256, 256, 0, stream>>>(readout_w, rwB, n4);
    }

    // encoder
    conv_first_gn_gelu<<<dim3(HID, NODES), 256, 0, stream>>>(
        x, conv_w_first, conv_b_first, gn_w_first, gn_b_first, actA);

    const int Ls[6] = {1000, 501, 251, 126, 64, 33};
    unsigned short* bufs[2] = {actA, actB};
    for (int i = 0; i < 4; ++i) {
        int Lin = Ls[i], Lout = Ls[i + 1];
        unsigned short* in  = bufs[i & 1];
        unsigned short* oup = bufs[(i + 1) & 1];
        gemm_conv<<<dim3((Lout + 31) / 32, HID / 128, NODES), 256, 0, stream>>>(
            in, wmidB + (size_t)i * HID * 1024, Ybuf, Lin, Lout, HID);
        gn_gelu<<<dim3(HID, NODES), 256, 0, stream>>>(
            Ybuf, conv_b_mid + i * HID, gn_w_mid + i * HID, gn_b_mid + i * HID,
            oup, nullptr, HID, Lout);
    }
    // last conv: in = actA (after 4 ping-pongs), Cout=64, Lout=33 -> latents
    gemm_conv<<<dim3((33 + 31) / 32, 1, NODES), 256, 0, stream>>>(
        actA, wlastB, Ybuf, 64, 33, ODIM);
    gn_gelu<<<dim3(ODIM, NODES), 256, 0, stream>>>(
        Ybuf, conv_b_last, gn_w_last, gn_b_last, latB, latF, ODIM, 33);

    // message passing
    for (int r = 0; r < NROUND; ++r) {
        gather_edges<<<NEDGE, 256, 0, stream>>>(latB, Ebuf);
        gemm_msg<<<dim3(DLAT / 32, NEDGE / 128), 256, 0, stream>>>(
            Ebuf, msgwB + (size_t)r * KMSG * DLAT, msg_b + r * DLAT, Mbuf);
        scatter_msgs<<<(NODES * DLAT + 255) / 256, 256, 0, stream>>>(Mbuf, latF, latB);
    }

    // readout
    reduce_nodes<<<(NB * DLAT + 255) / 256, 256, 0, stream>>>(latF, Zb);
    gemm_readout<<<(DLAT + 127) / 128, 256, 0, stream>>>(
        Zb, rwB, readout_b, (float*)d_out);
}
